// convCNP_47880295416556
// MI455X (gfx1250) — matrix operand order repaired and checked by reading, not compiled, not run on hardware
//
#include <hip/hip_runtime.h>
#include <hip/hip_bf16.h>

// ---------------------------------------------------------------------------
// ConvCNP forward for MI455X (gfx1250, wave32, WMMA).
//
// Layout strategy:
//  * all activations: f16, channel-interleaved (B, L+6, C) with 2-row zero
//    halo each side (+slack). im2col rows of a K=5 conv are then CONTIGUOUS,
//    so the 16x16x32 f16 WMMA A-fragment = two aligned v8h (one b128) loads.
//  * conv / dense weights: pre-transposed to (Cout, Rpad) f16, Rpad = K*Cin
//    rounded up to 32 (zero padded) -> B-fragment = two aligned v8h loads.
//  * dense1 reads the pooled activation buffer directly (halo folds into a
//    constant +256 half offset; flatten order (l,c) == our layout).
//  * 8 waves per workgroup: convs cover all cout-tiles (cache reuse of A
//    rows inside the WGP); dense1 uses split-K with an LDS reduction.
// ---------------------------------------------------------------------------

typedef __attribute__((ext_vector_type(8)))  _Float16 v8h;
typedef __attribute__((ext_vector_type(16))) _Float16 v16h;
typedef __attribute__((ext_vector_type(8)))  float    v8f;
typedef __attribute__((ext_vector_type(4)))  unsigned int v4u;

#define NB 16      // batch
#define NT 512     // time points
#define NM 8       // channels of y
#define NG 512     // grid points
#define BN_SCALE 0.99950037f   // 1/sqrt(1 + 1e-3)

// -------------------------------------------------------------- zero scratch
__global__ __launch_bounds__(256)
void zero_b128_kernel(v4u* __restrict__ p, int n16)
{
    int i = blockIdx.x * 256 + threadIdx.x;
    if (i < n16) p[i] = (v4u){};
}

// -------------------------------------------- pack weights (R,Cout)->(Cout,R)
__global__ __launch_bounds__(256)
void pack_wT_kernel(const float* __restrict__ w, _Float16* __restrict__ wT,
                    int Rreal, int Rpad, int Cout)
{
    int idx = blockIdx.x * 256 + threadIdx.x;
    if (idx >= Rpad * Cout) return;
    int co = idx / Rpad;
    int r  = idx - co * Rpad;
    float v = (r < Rreal) ? w[(size_t)r * Cout + co] : 0.0f;
    wT[idx] = (_Float16)v;
}

// ------------------------------------------------- RBF functional repr stage
// out channel c = 2*m + s ; s==0: density, s==1: conv / (density + 1e-8)
__global__ __launch_bounds__(128)
void func_repr_kernel(const float* __restrict__ x,     // (B,T)
                      const float* __restrict__ y,     // (B,T,M)
                      const int*   __restrict__ mask,  // (B,T,M)
                      const float* __restrict__ grid,  // (G)
                      const float* __restrict__ sigma, // (2)
                      _Float16*    __restrict__ act0)  // (B, G+6, 16) halo
{
    __shared__ float sx[64];
    __shared__ float sy[64][NM];
    __shared__ float sm[64][NM];

    const int b = blockIdx.x;
    const int g = blockIdx.y * 128 + threadIdx.x;
    const float gv  = grid[g];
    const float is0 = 1.0f / (sigma[0] * sigma[0]);
    const float is1 = 1.0f / (sigma[1] * sigma[1]);

    float accd[NM] = {0.f,0.f,0.f,0.f,0.f,0.f,0.f,0.f};
    float accc[NM] = {0.f,0.f,0.f,0.f,0.f,0.f,0.f,0.f};

    for (int t0 = 0; t0 < NT; t0 += 64) {
        __syncthreads();
        if (threadIdx.x < 64)
            sx[threadIdx.x] = x[b * NT + t0 + threadIdx.x];
        for (int i = threadIdx.x; i < 64 * NM; i += 128) {
            int tt = i >> 3, mm = i & 7;
            size_t gidx = ((size_t)b * NT + t0 + tt) * NM + mm;
            sy[tt][mm] = y[gidx];
            sm[tt][mm] = (float)mask[gidx];
        }
        __syncthreads();
        for (int t = 0; t < 64; ++t) {
            float d  = gv - sx[t];
            float q  = -0.5f * d * d;
            float e0 = __expf(q * is0);
            float e1 = __expf(q * is1);
            #pragma unroll
            for (int m = 0; m < NM; ++m) {
                accd[m] = fmaf(e0, sm[t][m], accd[m]);
                accc[m] = fmaf(e1, sy[t][m], accc[m]);
            }
        }
    }
    _Float16* out = act0 + ((size_t)b * (NG + 6) + g + 2) * 16;
    #pragma unroll
    for (int m = 0; m < NM; ++m) {
        float den = accd[m];
        out[2 * m]     = (_Float16)den;
        out[2 * m + 1] = (_Float16)(accc[m] / (den + 1e-8f));
    }
}

// ----------------------------------------------- conv-as-GEMM via f16 WMMA
// 8 waves / block; each wave computes one 16(pos) x 16(cout) tile.
// COTILES==4: waves cover 4 cout-tiles x 2 pos-tiles; COTILES==8: 8 x 1.
template<int RPAD, int RREAL, int COTILES>
__global__ __launch_bounds__(256)
void conv_wmma_kernel(const _Float16* __restrict__ actIn,  // (B, LinRows, Cin)
                      const _Float16* __restrict__ wT,     // (Cout, RPAD)
                      const float* __restrict__ bias,
                      const float* __restrict__ bnG,
                      const float* __restrict__ bnB,
                      _Float16* __restrict__ actOut,       // (B, LoutRows, Cout)
                      int Cin, int Cout, int LinRows, int LoutRows)
{
    const int wave = threadIdx.x >> 5;
    const int lane = threadIdx.x & 31;
    const int b    = blockIdx.x;
    const int pTile = (COTILES == 4) ? (blockIdx.y * 2 + (wave >> 2))
                                     : blockIdx.y;
    const int cTile = (COTILES == 4) ? (wave & 3) : wave;
    const int n  = lane & 15;
    const int hi = lane >> 4;

    // A row for lane = output position pTile*16 + n (M = lane % 16 for both
    // lane halves); contiguous im2col slice in the halo buffer.
    const _Float16* aBase = actIn + ((size_t)b * LinRows + pTile * 16 + n) * Cin;
    const _Float16* bBase = wT + (size_t)(cTile * 16 + n) * RPAD;

    v8f acc = {};
    #pragma unroll
    for (int kc = 0; kc < RPAD / 32; ++kc) {
        const int k0 = kc * 32;
        // A fragment: lane half hi -> K runs [k0+8*hi .. +7], [k0+16+8*hi .. +7]
        int r0 = k0 + hi * 8;
        int r1 = k0 + 16 + hi * 8;
        v8h a0 = {}; v8h a1 = {};
        if (r0 < RREAL) a0 = *(const v8h*)(aBase + r0);   // runs never straddle RREAL
        if (r1 < RREAL) a1 = *(const v8h*)(aBase + r1);   // (RREAL % 8 == 0)
        // B fragment: lane half hi -> K runs [k0+8*hi .. +7], [k0+16+8*hi .. +7]
        v8h b0 = *(const v8h*)(bBase + k0 + hi * 8);
        v8h b1 = *(const v8h*)(bBase + k0 + 16 + hi * 8);
        v16h a, bm;
        #pragma unroll
        for (int i = 0; i < 8; ++i) {
            a[i] = a0[i];  a[i + 8] = a1[i];
            bm[i] = b0[i]; bm[i + 8] = b1[i];
        }
        acc = __builtin_amdgcn_wmma_f32_16x16x32_f16(
                  false, a, false, bm, (short)0, acc, false, false);
    }

    // epilogue: bias + BN(scale folded) + ReLU -> f16 halo store
    const int co = cTile * 16 + n;
    const float g  = bnG[co] * BN_SCALE;
    const float bb = bnB[co];
    const float bs = bias[co];
    #pragma unroll
    for (int j = 0; j < 8; ++j) {
        int p = pTile * 16 + j + hi * 8;          // C/D: M = j + 8*hi, N = n
        float v = fmaf(acc[j] + bs, g, bb);
        v = fmaxf(v, 0.0f);
        actOut[((size_t)b * LoutRows + p + 2) * Cout + co] = (_Float16)v;
    }
}

// --------------------------------------------------------------- maxpool /2
__global__ __launch_bounds__(256)
void maxpool2_kernel(const _Float16* __restrict__ in, _Float16* __restrict__ out,
                     int Lout, int C, int LinRows, int LoutRows)
{
    int idx = blockIdx.x * 256 + threadIdx.x;
    if (idx >= NB * Lout * C) return;
    int c = idx % C;
    int l = (idx / C) % Lout;
    int b = idx / (C * Lout);
    float a = (float)in[((size_t)b * LinRows + 2 * l + 2) * C + c];
    float d = (float)in[((size_t)b * LinRows + 2 * l + 3) * C + c];
    out[((size_t)b * LoutRows + l + 2) * C + c] = (_Float16)fmaxf(a, d);
}

// ------------------- dense1: 16 x 16384 x 256, split-K WMMA + LDS reduction
// block = 8 waves; wave w handles K-chunks [w*64, w*64+64); LDS-reduce then
// fused bias + bn5 + relu.
__global__ __launch_bounds__(256)
void dense1_wmma_kernel(const _Float16* __restrict__ act6, // (16,134,128) halo
                        const _Float16* __restrict__ wT,   // (256, 16384)
                        const float* __restrict__ bias,
                        const float* __restrict__ bnG,
                        const float* __restrict__ bnB,
                        float* __restrict__ outZ)          // (16,256)
{
    __shared__ float part[8][256];
    const int cTile = blockIdx.x;
    const int wave  = threadIdx.x >> 5;
    const int lane  = threadIdx.x & 31;
    const int n  = lane & 15;
    const int hi = lane >> 4;

    // A row = batch n; flatten (l,c) == layout; halo => constant +256 offset
    const _Float16* aBase = act6 + (size_t)n * 134 * 128 + 256;
    const _Float16* bBase = wT + (size_t)(cTile * 16 + n) * 16384;

    v8f acc = {};
    const int kc0 = wave * 64;
    #pragma unroll 4
    for (int kc = kc0; kc < kc0 + 64; ++kc) {
        const int k0 = kc * 32;
        v8h a0 = *(const v8h*)(aBase + k0 + hi * 8);
        v8h a1 = *(const v8h*)(aBase + k0 + 16 + hi * 8);
        v8h b0 = *(const v8h*)(bBase + k0 + hi * 8);
        v8h b1 = *(const v8h*)(bBase + k0 + 16 + hi * 8);
        v16h a, bm;
        #pragma unroll
        for (int i = 0; i < 8; ++i) {
            a[i] = a0[i];  a[i + 8] = a1[i];
            bm[i] = b0[i]; bm[i + 8] = b1[i];
        }
        acc = __builtin_amdgcn_wmma_f32_16x16x32_f16(
                  false, a, false, bm, (short)0, acc, false, false);
    }

    #pragma unroll
    for (int j = 0; j < 8; ++j) part[wave][lane * 8 + j] = acc[j];
    __syncthreads();

    // slot t = lane*8 + j ; deterministic fixed-order 8-way sum
    const int t = threadIdx.x;
    float s = 0.0f;
    #pragma unroll
    for (int w = 0; w < 8; ++w) s += part[w][t];

    const int l2  = t >> 3, j = t & 7;
    const int n2  = l2 & 15, hi2 = l2 >> 4;
    const int bidx = j + hi2 * 8;                 // batch row (C/D M index)
    const int co   = cTile * 16 + n2;
    float v = fmaf(s + bias[co], bnG[co] * BN_SCALE, bnB[co]);
    outZ[bidx * 256 + co] = fmaxf(v, 0.0f);
}

// --------------------------------------------------- dense2 + sigmoid head
__global__ __launch_bounds__(256)
void dense2_kernel(const float* __restrict__ z,   // (16,256)
                   const float* __restrict__ w,   // (256)
                   const float* __restrict__ bias,// (1)
                   float* __restrict__ out)       // (16)
{
    __shared__ float red[256];
    const int b = blockIdx.x, t = threadIdx.x;
    red[t] = z[b * 256 + t] * w[t];
    __syncthreads();
    for (int s = 128; s > 0; s >>= 1) {
        if (t < s) red[t] += red[t + s];
        __syncthreads();
    }
    if (t == 0) out[b] = 1.0f / (1.0f + __expf(-(red[0] + bias[0])));
}

// ===========================================================================
extern "C" void kernel_launch(void* const* d_in, const int* in_sizes, int n_in,
                              void* d_out, int out_size, void* d_ws, size_t ws_size,
                              hipStream_t stream)
{
    const float* x     = (const float*)d_in[0];
    const float* y     = (const float*)d_in[1];
    const int*   mask  = (const int*)  d_in[2];
    const float* grid  = (const float*)d_in[3];
    const float* sigma = (const float*)d_in[4];
    const float* c1w = (const float*)d_in[5];  const float* c1b = (const float*)d_in[6];
    const float* c2w = (const float*)d_in[7];  const float* c2b = (const float*)d_in[8];
    const float* c3w = (const float*)d_in[9];  const float* c3b = (const float*)d_in[10];
    const float* c4w = (const float*)d_in[11]; const float* c4b = (const float*)d_in[12];
    const float* d1w = (const float*)d_in[13]; const float* d1b = (const float*)d_in[14];
    const float* d2w = (const float*)d_in[15]; const float* d2b = (const float*)d_in[16];
    const float* bn1g = (const float*)d_in[17]; const float* bn1b = (const float*)d_in[18];
    const float* bn2g = (const float*)d_in[19]; const float* bn2b = (const float*)d_in[20];
    const float* bn3g = (const float*)d_in[21]; const float* bn3b = (const float*)d_in[22];
    const float* bn4g = (const float*)d_in[23]; const float* bn4b = (const float*)d_in[24];
    const float* bn5g = (const float*)d_in[25]; const float* bn5b = (const float*)d_in[26];

    // ---- workspace carve-up (256B aligned; activation blocks contiguous) ----
    char* ws = (char*)d_ws;
    size_t off = 0;
    auto take = [&](size_t bytes) -> char* {
        char* p = ws + off;
        off += (bytes + 255) & ~(size_t)255;
        return p;
    };
    _Float16* act0 = (_Float16*)take((size_t)NB * 518 *  16 * 2); // RBF out
    _Float16* act1 = (_Float16*)take((size_t)NB * 518 *  64 * 2); // conv1
    _Float16* act2 = (_Float16*)take((size_t)NB * 518 *  64 * 2); // conv2
    _Float16* act3 = (_Float16*)take((size_t)NB * 262 *  64 * 2); // pool1
    _Float16* act4 = (_Float16*)take((size_t)NB * 262 * 128 * 2); // conv3
    _Float16* act5 = (_Float16*)take((size_t)NB * 262 * 128 * 2); // conv4
    _Float16* act6 = (_Float16*)take((size_t)NB * 134 * 128 * 2); // pool2
    _Float16* w1T  = (_Float16*)take((size_t)64  *    96 * 2);
    _Float16* w2T  = (_Float16*)take((size_t)64  *   320 * 2);
    _Float16* w3T  = (_Float16*)take((size_t)128 *   320 * 2);
    _Float16* w4T  = (_Float16*)take((size_t)128 *   640 * 2);
    _Float16* d1T  = (_Float16*)take((size_t)256 * 16384 * 2);
    float*    dz   = (float*)   take((size_t)NB * 256 * 4);

    // 1. zero all activation buffers (halos must be 0; all sizes 256B multiples)
    {
        int n16 = (int)(((size_t)NB * (518*16 + 518*64 + 518*64 + 262*64 +
                                       262*128 + 262*128 + 134*128) * 2) / 16);
        zero_b128_kernel<<<(n16 + 255) / 256, 256, 0, stream>>>((v4u*)act0, n16);
    }

    // 2. pack weights to (Cout, Rpad) f16, zero-padded
    pack_wT_kernel<<<(  96 *  64 + 255) / 256, 256, 0, stream>>>(c1w, w1T,    80,    96,  64);
    pack_wT_kernel<<<( 320 *  64 + 255) / 256, 256, 0, stream>>>(c2w, w2T,   320,   320,  64);
    pack_wT_kernel<<<( 320 * 128 + 255) / 256, 256, 0, stream>>>(c3w, w3T,   320,   320, 128);
    pack_wT_kernel<<<( 640 * 128 + 255) / 256, 256, 0, stream>>>(c4w, w4T,   640,   640, 128);
    pack_wT_kernel<<<(16384 * 256 + 255) / 256, 256, 0, stream>>>(d1w, d1T, 16384, 16384, 256);

    // 3. RBF functional representation -> act0 (B, 518, 16)
    func_repr_kernel<<<dim3(NB, NG / 128), 128, 0, stream>>>(
        x, y, mask, grid, sigma, act0);

    // 4. conv stack (8 waves/block; each wave one 16x16 WMMA tile)
    conv_wmma_kernel< 96,  80, 4><<<dim3(NB, 16), 256, 0, stream>>>(
        act0, w1T, c1b, bn1g, bn1b, act1,  16,  64, 518, 518);
    conv_wmma_kernel<320, 320, 4><<<dim3(NB, 16), 256, 0, stream>>>(
        act1, w2T, c2b, bn2g, bn2b, act2,  64,  64, 518, 518);
    maxpool2_kernel<<<(NB * 256 * 64 + 255) / 256, 256, 0, stream>>>(
        act2, act3, 256, 64, 518, 262);
    conv_wmma_kernel<320, 320, 8><<<dim3(NB, 16), 256, 0, stream>>>(
        act3, w3T, c3b, bn3g, bn3b, act4,  64, 128, 262, 262);
    conv_wmma_kernel<640, 640, 8><<<dim3(NB, 16), 256, 0, stream>>>(
        act4, w4T, c4b, bn4g, bn4b, act5, 128, 128, 262, 262);
    maxpool2_kernel<<<(NB * 128 * 128 + 255) / 256, 256, 0, stream>>>(
        act5, act6, 128, 128, 262, 134);

    // 5. dense1 (16 x 16384 x 256) split-K + LDS reduce + bn5 + relu
    dense1_wmma_kernel<<<16, 256, 0, stream>>>(act6, d1T, d1b, bn5g, bn5b, dz);

    // 6. dense2 + sigmoid -> d_out (16 floats)
    dense2_kernel<<<NB, 256, 0, stream>>>(dz, d2w, d2b, (float*)d_out);

    (void)in_sizes; (void)n_in; (void)out_size; (void)ws_size;
}
